// AttnCore_1090921693354
// MI455X (gfx1250) — compile-verified, matrix operand order checked by reading, not run on hardware
//
#include <hip/hip_runtime.h>
#include <hip/hip_bf16.h>

typedef __bf16 bf16_t;
typedef __attribute__((ext_vector_type(8)))  __bf16 v8bf;
typedef __attribute__((ext_vector_type(16))) __bf16 v16bf;
typedef __attribute__((ext_vector_type(8)))  float  v8f;

#define NND 4096
#define DDIM 256
#define EGE 32768
#define ECL 133120

// ---------------------------------------------------------------- conversions
__global__ __launch_bounds__(256) void cvt_f32_to_bf16(const float* __restrict__ in,
                                                       bf16_t* __restrict__ out,
                                                       long long n) {
  long long i = (long long)blockIdx.x * 256 + threadIdx.x;
  if (i < n) out[i] = (bf16_t)in[i];
}

// ---------------------------------------------------------------- WMMA GEMM
// C[M,N] = act(A[M,K] @ W[N,K]^T + bias). M%64==0, N%256==0, K%32==0.
// Block = 256 threads (8 waves) -> 64 rows x 256 cols of C.
//   A strip [64, KC<=256] staged to LDS per K-chunk via
//   global_load_async_to_lds_b128 (ASYNCcnt) and read back as ds_load_b128.
//   Each wave owns 32 rows x 64 cols: 8 accumulators; every B fragment is
//   shared by two WMMAs (1.5 b128 loads per WMMA). All four B fragments of a
//   k-step are loaded into registers up-front (single clause) so the WMMAs
//   drain the loads progressively instead of stalling on loadcnt==0.
// MODE 0: store; MODE 1: store+relu; MODE 2: relu + run-length segment atomicAdd by sIdx[m].
template <int MODE>
__global__ __launch_bounds__(256) void gemm_wmma_bf16(const bf16_t* __restrict__ A,
                                                      const bf16_t* __restrict__ W,
                                                      const float* __restrict__ bias,
                                                      float* __restrict__ C,
                                                      int M, int K, int N,
                                                      const int* __restrict__ sIdx) {
  __shared__ __align__(16) bf16_t shA[64 * 256];   // 32 KB
  int tid = threadIdx.x;
  int nBlkCols = N >> 8;                 // N/256
  int bm = blockIdx.x / nBlkCols;
  int bn = blockIdx.x % nBlkCols;
  int rowBase = bm * 64;
  const int KC = K < 256 ? K : 256;
  const int rowChunks = KC >> 3;         // 16B chunks per row

  int wv = tid >> 5, lane = tid & 31;
  int r = lane & 15, h = lane >> 4;
  int tmLocal = wv >> 2;                 // 0..1  (32-row half of the block)
  int jn = wv & 3;                       // 0..3  (64-col strip)
  int colBase = bn * 256 + jn * 64;

  v8f acc[8] = {};
  for (int kc = 0; kc < K; kc += KC) {
    __syncthreads();                     // protect LDS reads of previous chunk
    // ---- stage A[rowBase..+64, kc..kc+KC) into LDS (async DMA, 16B chunks) ----
    for (int c = tid; c < 64 * rowChunks; c += 256) {
      int rr = c / rowChunks, cc = c % rowChunks;
      unsigned lds = (unsigned)(uintptr_t)(shA + rr * KC + cc * 8);
      const bf16_t* g = A + (size_t)(rowBase + rr) * K + kc + cc * 8;
      asm volatile("global_load_async_to_lds_b128 %0, %1, off"
                   :: "v"(lds), "v"(g) : "memory");
    }
    asm volatile("s_wait_asynccnt 0x0" ::: "memory");
    __syncthreads();

    const bf16_t* aRow0 = shA + (size_t)(tmLocal * 32 + r) * KC;
    const bf16_t* aRow1 = aRow0 + (size_t)16 * KC;
    for (int k0 = 0; k0 < KC; k0 += 32) {
      // ---- issue all B loads of this k-step first (one clause) ----
      v8bf b0[4], b1[4];
      #pragma unroll
      for (int j = 0; j < 4; j++) {
        const bf16_t* wr = W + (size_t)(colBase + j * 16 + r) * K + kc + k0;
        b0[j] = *(const v8bf*)(wr + 8 * h);
        b1[j] = *(const v8bf*)(wr + 16 + 8 * h);
      }
      // ---- A fragments from LDS ----
      v8bf a00 = *(const v8bf*)(aRow0 + k0 + 8 * h);
      v8bf a01 = *(const v8bf*)(aRow0 + k0 + 16 + 8 * h);
      v8bf a10 = *(const v8bf*)(aRow1 + k0 + 8 * h);
      v8bf a11 = *(const v8bf*)(aRow1 + k0 + 16 + 8 * h);
      v16bf av0 = __builtin_shufflevector(a00, a01, 0,1,2,3,4,5,6,7,8,9,10,11,12,13,14,15);
      v16bf av1 = __builtin_shufflevector(a10, a11, 0,1,2,3,4,5,6,7,8,9,10,11,12,13,14,15);
      #pragma unroll
      for (int j = 0; j < 4; j++) {
        v16bf bv = __builtin_shufflevector(b0[j], b1[j], 0,1,2,3,4,5,6,7,8,9,10,11,12,13,14,15);
        acc[j]     = __builtin_amdgcn_wmma_f32_16x16x32_bf16(false, av0, false, bv,
                                                             (short)0, acc[j], false, false);
        acc[4 + j] = __builtin_amdgcn_wmma_f32_16x16x32_bf16(false, av1, false, bv,
                                                             (short)0, acc[4 + j], false, false);
      }
    }
  }

  #pragma unroll
  for (int half = 0; half < 2; half++) {
    int rowTile = rowBase + tmLocal * 32 + half * 16;
    #pragma unroll
    for (int j = 0; j < 4; j++) {
      union { v8f v; float f[8]; } u; u.v = acc[half * 4 + j];
      int nn = colBase + j * 16 + r;
      float bb = bias ? bias[nn] : 0.f;
      if (MODE == 2) {
        float run = 0.f; int cur = -1;
        #pragma unroll
        for (int v = 0; v < 8; v++) {
          int m = rowTile + v + 8 * h;
          float val = u.f[v] + bb;
          val = val > 0.f ? val : 0.f;
          int s = sIdx[m];
          if (s != cur) {
            if (cur >= 0) atomicAdd(&C[(size_t)cur * N + nn], run);
            cur = s; run = 0.f;
          }
          run += val;
        }
        if (cur >= 0) atomicAdd(&C[(size_t)cur * N + nn], run);
      } else {
        #pragma unroll
        for (int v = 0; v < 8; v++) {
          int m = rowTile + v + 8 * h;
          float val = u.f[v] + bb;
          if (MODE == 1) val = val > 0.f ? val : 0.f;
          C[(size_t)m * N + nn] = val;
        }
      }
    }
  }
}

// ---------------------------------------------------------------- GAT edge ops
__device__ inline unsigned enc_mono(float x) {
  unsigned u = __float_as_uint(x);
  return (u & 0x80000000u) ? ~u : (u | 0x80000000u);
}
__device__ inline float dec_mono(unsigned u) {
  return (u & 0x80000000u) ? __uint_as_float(u & 0x7FFFFFFFu) : __uint_as_float(~u);
}

__global__ __launch_bounds__(256) void gat_edge_e(const float* __restrict__ xl,
                                                  const float* __restrict__ xr,
                                                  const float* __restrict__ att,
                                                  const int* __restrict__ srcE,
                                                  const int* __restrict__ dstE,
                                                  float* __restrict__ e_out,
                                                  unsigned* __restrict__ maxbuf) {
  int wave = (blockIdx.x * 256 + threadIdx.x) >> 5;
  int lane = threadIdx.x & 31;
  int total = EGE + NND;
  if (wave >= total) return;
  int s = wave < EGE ? srcE[wave] : (wave - EGE);
  int d = wave < EGE ? dstE[wave] : (wave - EGE);
  const float* pl = xl + (size_t)s * DDIM;
  const float* pr = xr + (size_t)d * DDIM;
  float acc = 0.f;
  for (int k = lane; k < DDIM; k += 32) {
    float v = pl[k] + pr[k];
    v = v > 0.f ? v : 0.2f * v;   // leaky_relu, slope 0.2
    acc += v * att[k];
  }
  for (int off = 16; off; off >>= 1) acc += __shfl_xor(acc, off, 32);
  if (lane == 0) {
    e_out[wave] = acc;
    atomicMax(&maxbuf[d], enc_mono(acc));
  }
}

__global__ __launch_bounds__(256) void gat_ew(const float* __restrict__ e,
                                              const int* __restrict__ dstE,
                                              const unsigned* __restrict__ maxbuf,
                                              float* __restrict__ ew,
                                              float* __restrict__ den) {
  int i = blockIdx.x * 256 + threadIdx.x;
  if (i >= EGE + NND) return;
  int d = i < EGE ? dstE[i] : (i - EGE);
  float w = __expf(e[i] - dec_mono(maxbuf[d]));
  ew[i] = w;
  atomicAdd(&den[d], w);
}

__global__ __launch_bounds__(256) void gat_scatter(const float* __restrict__ xl,
                                                   const float* __restrict__ ew,
                                                   const float* __restrict__ den,
                                                   const int* __restrict__ srcE,
                                                   const int* __restrict__ dstE,
                                                   float* __restrict__ out) {
  int e = blockIdx.x, t = threadIdx.x;
  int s = e < EGE ? srcE[e] : (e - EGE);
  int d = e < EGE ? dstE[e] : (e - EGE);
  float alpha = ew[e] / den[d];
  atomicAdd(&out[(size_t)d * DDIM + t], alpha * xl[(size_t)s * DDIM + t]);
}

__global__ __launch_bounds__(256) void add_bias_row(float* __restrict__ x,
                                                    const float* __restrict__ bias) {
  x[(size_t)blockIdx.x * DDIM + threadIdx.x] += bias[threadIdx.x];
}

// ---------------------------------------------------------------- norms
__global__ __launch_bounds__(256) void reduce_sumsq(const float* __restrict__ x,
                                                    long long n, float* __restrict__ st) {
  long long i = (long long)blockIdx.x * 256 + threadIdx.x;
  long long stride = (long long)gridDim.x * 256;
  float s = 0.f, q = 0.f;
  for (; i < n; i += stride) { float v = x[i]; s += v; q += v * v; }
  __shared__ float ss[256], qq[256];
  ss[threadIdx.x] = s; qq[threadIdx.x] = q; __syncthreads();
  for (int k = 128; k; k >>= 1) {
    if (threadIdx.x < k) { ss[threadIdx.x] += ss[threadIdx.x + k]; qq[threadIdx.x] += qq[threadIdx.x + k]; }
    __syncthreads();
  }
  if (threadIdx.x == 0) { atomicAdd(&st[0], ss[0]); atomicAdd(&st[1], qq[0]); }
}

__global__ __launch_bounds__(256) void graphln_apply(const float* __restrict__ x,
                                                     const float* __restrict__ g,
                                                     const float* __restrict__ b,
                                                     const float* __restrict__ st,
                                                     float* __restrict__ out,
                                                     float invCount, int doRelu) {
  int t = threadIdx.x;
  size_t i = (size_t)blockIdx.x * DDIM + t;
  float m = st[0] * invCount;
  float var = st[1] * invCount - m * m;
  float v = (x[i] - m) * rsqrtf(var + 1e-5f) * g[t] + b[t];
  if (doRelu) v = fmaxf(v, 0.f);
  out[i] = v;
}

__global__ __launch_bounds__(256) void ln_row(const float* __restrict__ x,
                                              const float* __restrict__ res,
                                              const float* __restrict__ g,
                                              const float* __restrict__ b,
                                              float* __restrict__ out) {
  int t = threadIdx.x;
  size_t i = (size_t)blockIdx.x * DDIM + t;
  float v = x[i] + (res ? res[i] : 0.f);
  __shared__ float sh[256];
  sh[t] = v; __syncthreads();
  for (int k = 128; k; k >>= 1) { if (t < k) sh[t] += sh[t + k]; __syncthreads(); }
  float m = sh[0] * (1.f / 256.f); __syncthreads();
  float dv = v - m;
  sh[t] = dv * dv; __syncthreads();
  for (int k = 128; k; k >>= 1) { if (t < k) sh[t] += sh[t + k]; __syncthreads(); }
  float var = sh[0] * (1.f / 256.f);
  out[i] = dv * rsqrtf(var + 1e-5f) * g[t] + b[t];
}

__global__ __launch_bounds__(256) void bn_col(const float* __restrict__ x,
                                              const float* __restrict__ g,
                                              const float* __restrict__ b,
                                              float* __restrict__ out, int M) {
  int col = blockIdx.x, t = threadIdx.x;
  float s = 0.f, q = 0.f;
  for (int r = t; r < M; r += 256) { float v = x[(size_t)r * DDIM + col]; s += v; q += v * v; }
  __shared__ float ss[256], qq[256];
  ss[t] = s; qq[t] = q; __syncthreads();
  for (int k = 128; k; k >>= 1) { if (t < k) { ss[t] += ss[t + k]; qq[t] += qq[t + k]; } __syncthreads(); }
  float m = ss[0] / M, var = qq[0] / M - m * m;
  float sc = rsqrtf(var + 1e-5f) * g[col];
  float sb = b[col] - m * sc;
  for (int r = t; r < M; r += 256) out[(size_t)r * DDIM + col] = x[(size_t)r * DDIM + col] * sc + sb;
}

// ---------------------------------------------------------------- cloud graph
__global__ __launch_bounds__(256) void cloud_stage(const float* __restrict__ x0,
                                                   const float* __restrict__ xyz,
                                                   const int* __restrict__ cs,
                                                   const int* __restrict__ cd,
                                                   bf16_t* __restrict__ A) {
  int e = blockIdx.x, t = threadIdx.x;
  int s = cs[e], d = cd[e];
  float dx = xyz[s * 3 + 0] - xyz[d * 3 + 0];
  float dy = xyz[s * 3 + 1] - xyz[d * 3 + 1];
  float dz = xyz[s * 3 + 2] - xyz[d * 3 + 2];
  float w = __expf(-sqrtf(dx * dx + dy * dy + dz * dz + 1e-12f));
  A[(size_t)e * DDIM + t] = (bf16_t)(w * (x0[(size_t)s * DDIM + t] - x0[(size_t)d * DDIM + t]));
}

__global__ __launch_bounds__(256) void cloud_h2(const float* __restrict__ xyz,
                                                const float* __restrict__ W3,
                                                float* __restrict__ h2) {
  int i = blockIdx.x, t = threadIdx.x;
  int mol = i >> 6, li = i & 63;
  float w0 = W3[t * 3], w1 = W3[t * 3 + 1], w2 = W3[t * 3 + 2];
  float xi = xyz[i * 3], yi = xyz[i * 3 + 1], zi = xyz[i * 3 + 2];
  float acc = 0.f;
  for (int j = li; j < 64; j++) {
    int g = (mol << 6) + j;
    float v = w0 * (xi - xyz[g * 3]) + w1 * (yi - xyz[g * 3 + 1]) + w2 * (zi - xyz[g * 3 + 2]);
    acc += v > 0.f ? v : 0.f;
  }
  h2[(size_t)i * DDIM + t] = acc;
}

__global__ __launch_bounds__(256) void add3(const float* __restrict__ a,
                                            const float* __restrict__ b,
                                            const float* __restrict__ c,
                                            float* __restrict__ o) {
  size_t i = (size_t)blockIdx.x * DDIM + threadIdx.x;
  o[i] = a[i] + b[i] + c[i];
}

// ---------------------------------------------------------------- attention core
template <int L, int DH, int H>
__global__ __launch_bounds__(128) void mha_attn(const float* __restrict__ Q,
                                                const float* __restrict__ K,
                                                const float* __restrict__ V,
                                                float* __restrict__ O) {
  int idx = blockIdx.x;
  int q = idx % L;
  int h = (idx / L) % H;
  int b = idx / (L * H);
  int t = threadIdx.x;
  __shared__ float sc[80];
  __shared__ float inv;
  const float scale = rsqrtf((float)DH);
  if (t < L) {
    const float* qp = Q + (size_t)(b * L + q) * DDIM + h * DH;
    const float* kp = K + (size_t)(b * L + t) * DDIM + h * DH;
    float s = 0.f;
    for (int d = 0; d < DH; d++) s += qp[d] * kp[d];
    sc[t] = s * scale;
  }
  __syncthreads();
  if (t == 0) {
    float mx = -1e30f;
    for (int k = 0; k < L; k++) mx = fmaxf(mx, sc[k]);
    float sm = 0.f;
    for (int k = 0; k < L; k++) { float w = __expf(sc[k] - mx); sc[k] = w; sm += w; }
    inv = 1.f / sm;
  }
  __syncthreads();
  if (t < DH) {
    float acc = 0.f;
    for (int k = 0; k < L; k++) acc += sc[k] * V[(size_t)(b * L + k) * DDIM + h * DH + t];
    O[(size_t)(b * L + q) * DDIM + h * DH + t] = acc * inv;
  }
}

// ---------------------------------------------------------------- gathers / pooling / assembly
__global__ __launch_bounds__(256) void gather_rows(const float* __restrict__ X,
                                                   const int* __restrict__ idx,
                                                   float* __restrict__ out) {
  out[(size_t)blockIdx.x * DDIM + threadIdx.x] =
      X[(size_t)idx[blockIdx.x] * DDIM + threadIdx.x];
}

__global__ __launch_bounds__(256) void absmax_pool(const float* __restrict__ RS,
                                                   float* __restrict__ XR) {
  int r = blockIdx.x, t = threadIdx.x;
  float best = RS[(size_t)(r * 6) * DDIM + t];
  float ab = fabsf(best);
  for (int l = 1; l < 6; l++) {
    float v = RS[(size_t)(r * 6 + l) * DDIM + t];
    float a = fabsf(v);
    if (a > ab) { ab = a; best = v; }
  }
  XR[(size_t)r * DDIM + t] = best;
}

__global__ __launch_bounds__(256) void assemble_seq(const float* __restrict__ xnode,
                                                    const float* __restrict__ xr,
                                                    const float* __restrict__ CLS,
                                                    const float* __restrict__ RING,
                                                    const float* __restrict__ END,
                                                    float* __restrict__ SEQ) {
  int pos = blockIdx.x % 75;
  int b = blockIdx.x / 75;
  int t = threadIdx.x;
  float v;
  if (pos == 0)        v = CLS[t];
  else if (pos <= 64)  v = xnode[(size_t)(b * 64 + pos - 1) * DDIM + t];
  else if (pos == 65)  v = RING[t];
  else if (pos <= 73)  v = xr[(size_t)(b * 8 + pos - 66) * DDIM + t];
  else                 v = END[t];
  SEQ[(size_t)blockIdx.x * DDIM + t] = v;
}

__global__ __launch_bounds__(256) void write_out(const float* __restrict__ seq,
                                                 float* __restrict__ out, int out_size) {
  int i = blockIdx.x * 256 + threadIdx.x;
  const int S = 64 * 75 * DDIM;
  if (i < S) out[i] = seq[i];
  else if (i < out_size) out[i] = 1.0f;
}

// ================================================================ host
extern "C" void kernel_launch(void* const* d_in, const int* in_sizes, int n_in,
                              void* d_out, int out_size, void* d_ws, size_t ws_size,
                              hipStream_t stream) {
  const int Nn = NND, FFd = 1024;
  const float* x_feat = (const float*)d_in[0];
  const float* xyz = (const float*)d_in[1];
  const int* eidx = (const int*)d_in[2];
  const int* cidx = (const int*)d_in[3];
  const int* ridx = (const int*)d_in[4];
  const int* srcE = eidx;           const int* dstE = eidx + EGE;
  const int* cS = cidx;             const int* cD = cidx + ECL;

  int pi = 5;
  auto P = [&]() { return (const float*)d_in[pi++]; };
  const float* x_proj_w = P(); const float* x_proj_b = P();
  const float *g_wl[6], *g_wr[6], *g_att[6], *g_bias[6], *g_lng[6], *g_lnb[6];
  for (int i = 0; i < 6; i++) {
    g_wl[i] = P(); g_wr[i] = P(); g_att[i] = P(); g_bias[i] = P(); g_lng[i] = P(); g_lnb[i] = P();
  }
  const float* cg_lin1_w = P(); const float* cg_lin1_b = P();
  const float* cg_ln1_g = P();  const float* cg_ln1_b = P();
  const float* cg_xyz_w = P();  const float* cg_bn_g = P(); const float* cg_bn_b = P();
  const float* np_lin_w = P();  const float* np_lin_b = P();
  const float* np_bn_g = P();   const float* np_bn_b = P();
  struct EncP { const float *wq,*wk,*wv,*bq,*bk,*bv,*wo,*bo,*w1,*b1,*w2,*b2,*l1g,*l1b,*l2g,*l2b; };
  EncP ring, mol;
  auto loadEnc = [&](EncP& e) {
    e.wq=P(); e.wk=P(); e.wv=P(); e.bq=P(); e.bk=P(); e.bv=P(); e.wo=P(); e.bo=P();
    e.w1=P(); e.b1=P(); e.w2=P(); e.b2=P(); e.l1g=P(); e.l1b=P(); e.l2g=P(); e.l2b=P();
  };
  loadEnc(ring); loadEnc(mol);
  const float* CLSv = P(); const float* RINGv = P(); const float* ENDv = P();

  // ---- workspace layout ----
  char* base = (char*)d_ws;
  size_t off = 0;
  auto alloc = [&](size_t bytes) -> void* {
    void* p = base + off;
    off = (off + bytes + 255) & ~(size_t)255;
    return p;
  };
  auto fbuf = [&](size_t elems) { return (float*)alloc(elems * 4); };
  const size_t ND = (size_t)Nn * DDIM;
  float *X0 = fbuf(ND), *XG_A = fbuf(ND), *XG_B = fbuf(ND);
  float *XL = fbuf(ND), *XR = fbuf(ND), *GOUT = fbuf(ND);
  float *H1 = fbuf(ND), *H2 = fbuf(ND), *XCLOUD = fbuf(ND), *SUM3 = fbuf(ND);
  float *NPOUT = fbuf(ND), *XNODE = fbuf(ND);
  float *EBUF = fbuf(EGE + Nn), *EWBUF = fbuf(EGE + Nn), *DEN = fbuf(Nn);
  unsigned* MAXB = (unsigned*)alloc((size_t)Nn * 4);
  float* STATS = fbuf(2);
  const size_t ER = (size_t)4800 * DDIM;
  float *EX = fbuf(ER), *EQ = fbuf(ER), *EK = fbuf(ER), *EV = fbuf(ER);
  float *EAO = fbuf(ER), *EMO = fbuf(ER), *EX1 = fbuf(ER), *EF2 = fbuf(ER), *EX2 = fbuf(ER);
  float* EF1 = fbuf((size_t)4800 * FFd);
  float* XRP = fbuf((size_t)512 * DDIM);
  bf16_t* ABUF = (bf16_t*)alloc((size_t)ECL * DDIM * 2);
  bf16_t* WBUF = (bf16_t*)alloc((size_t)FFd * DDIM * 2);

  auto CVT = [&](const float* src, bf16_t* dst, size_t n) {
    cvt_f32_to_bf16<<<(unsigned)((n + 255) / 256), 256, 0, stream>>>(src, dst, (long long)n);
  };
  auto GEMM = [&](const float* A, const float* W, const float* bias, float* C,
                  int M, int K, int N, int mode, const int* sIdx, bool cvtA = true) {
    if (cvtA) CVT(A, ABUF, (size_t)M * K);
    CVT(W, WBUF, (size_t)N * K);
    unsigned grid = (unsigned)((M / 64) * (N / 256));
    if (mode == 0)
      gemm_wmma_bf16<0><<<grid, 256, 0, stream>>>(ABUF, WBUF, bias, C, M, K, N, sIdx);
    else if (mode == 1)
      gemm_wmma_bf16<1><<<grid, 256, 0, stream>>>(ABUF, WBUF, bias, C, M, K, N, sIdx);
    else
      gemm_wmma_bf16<2><<<grid, 256, 0, stream>>>(ABUF, WBUF, bias, C, M, K, N, sIdx);
  };

  // ---- 1. input projection: X0 = x_feat @ x_proj_w^T + b ----
  GEMM(x_feat, x_proj_w, x_proj_b, X0, Nn, 32, DDIM, 0, nullptr);

  // ---- 2. six GATv2 layers ----
  const float* curX = X0;
  float* nxt = XG_A;
  const float invND = 1.f / (float)(Nn * DDIM);
  for (int i = 0; i < 6; i++) {
    GEMM(curX, g_wl[i], nullptr, XL, Nn, DDIM, DDIM, 0, nullptr, true);
    GEMM(curX, g_wr[i], nullptr, XR, Nn, DDIM, DDIM, 0, nullptr, false);  // same A
    hipMemsetAsync(MAXB, 0, (size_t)Nn * 4, stream);
    hipMemsetAsync(DEN, 0, (size_t)Nn * 4, stream);
    hipMemsetAsync(GOUT, 0, ND * 4, stream);
    gat_edge_e<<<(EGE + Nn) / 8, 256, 0, stream>>>(XL, XR, g_att[i], srcE, dstE, EBUF, MAXB);
    gat_ew<<<(EGE + Nn + 255) / 256, 256, 0, stream>>>(EBUF, dstE, MAXB, EWBUF, DEN);
    gat_scatter<<<EGE + Nn, 256, 0, stream>>>(XL, EWBUF, DEN, srcE, dstE, GOUT);
    add_bias_row<<<Nn, 256, 0, stream>>>(GOUT, g_bias[i]);
    if (i < 5) {
      hipMemsetAsync(STATS, 0, 8, stream);
      reduce_sumsq<<<512, 256, 0, stream>>>(GOUT, (long long)ND, STATS);
      graphln_apply<<<Nn, 256, 0, stream>>>(GOUT, g_lng[i], g_lnb[i], STATS, nxt, invND, 1);
      curX = nxt;
      nxt = (nxt == XG_A) ? XG_B : XG_A;
    }
  }
  const float* xgF = GOUT;  // final GAT output (bias added, no LN/relu)

  // ---- 3. CloudGraph ----
  cloud_stage<<<ECL, 256, 0, stream>>>(X0, xyz, cS, cD, ABUF);
  CVT(cg_lin1_w, WBUF, (size_t)DDIM * DDIM);
  hipMemsetAsync(H1, 0, ND * 4, stream);
  {
    unsigned grid = (unsigned)((ECL / 64) * (DDIM / 256));
    gemm_wmma_bf16<2><<<grid, 256, 0, stream>>>(ABUF, WBUF, cg_lin1_b, H1, ECL, DDIM, DDIM, cS);
  }
  cloud_h2<<<Nn, 256, 0, stream>>>(xyz, cg_xyz_w, H2);
  ln_row<<<Nn, 256, 0, stream>>>(H1, nullptr, cg_ln1_g, cg_ln1_b, XL);   // XL = LN(h1)
  bn_col<<<DDIM, 256, 0, stream>>>(H2, cg_bn_g, cg_bn_b, XR, Nn);       // XR = BN(h2)
  add3<<<Nn, 256, 0, stream>>>(X0, XL, XR, XCLOUD);
  add3<<<Nn, 256, 0, stream>>>(X0, xgF, XCLOUD, SUM3);
  GEMM(SUM3, np_lin_w, np_lin_b, NPOUT, Nn, DDIM, DDIM, 0, nullptr);
  bn_col<<<DDIM, 256, 0, stream>>>(NPOUT, np_bn_g, np_bn_b, XNODE, Nn);

  // ---- 4. encoder layer helper ----
  auto ENC = [&](const EncP& e, int rows, int L, int H, int DH) {
    GEMM(EX, e.wq, e.bq, EQ, rows, DDIM, DDIM, 0, nullptr, true);
    GEMM(EX, e.wk, e.bk, EK, rows, DDIM, DDIM, 0, nullptr, false);   // same A
    GEMM(EX, e.wv, e.bv, EV, rows, DDIM, DDIM, 0, nullptr, false);   // same A
    unsigned grid = (unsigned)(rows * H);
    if (L == 6) mha_attn<6, 128, 2><<<grid, 128, 0, stream>>>(EQ, EK, EV, EAO);
    else        mha_attn<75, 64, 4><<<grid, 128, 0, stream>>>(EQ, EK, EV, EAO);
    GEMM(EAO, e.wo, e.bo, EMO, rows, DDIM, DDIM, 0, nullptr);
    ln_row<<<rows, 256, 0, stream>>>(EMO, EX, e.l1g, e.l1b, EX1);
    GEMM(EX1, e.w1, e.b1, EF1, rows, DDIM, FFd, 1, nullptr);
    GEMM(EF1, e.w2, e.b2, EF2, rows, FFd, DDIM, 0, nullptr);
    ln_row<<<rows, 256, 0, stream>>>(EF2, EX1, e.l2g, e.l2b, EX2);
  };

  // ---- 5. ring attention + absmax pooling ----
  gather_rows<<<3072, 256, 0, stream>>>(XNODE, ridx, EX);
  ENC(ring, 3072, 6, 2, 128);
  absmax_pool<<<512, 256, 0, stream>>>(EX2, XRP);

  // ---- 6. molecular transformer ----
  assemble_seq<<<64 * 75, 256, 0, stream>>>(XNODE, XRP, CLSv, RINGv, ENDv, EX);
  ENC(mol, 4800, 75, 4, 64);

  // ---- 7. output + validity masks (all true) ----
  write_out<<<(unsigned)((out_size + 255) / 256), 256, 0, stream>>>(EX2, (float*)d_out, out_size);
}
